// GATv2_43499428773954
// MI455X (gfx1250) — compile-verified, matrix operand order checked by reading, not run on hardware
//
#include <hip/hip_runtime.h>

typedef __attribute__((ext_vector_type(16))) _Float16 v16h;
typedef __attribute__((ext_vector_type(8)))  _Float16 v8h;
typedef __attribute__((ext_vector_type(8)))  float    v8f;

#define G_DIM 5000
#define IN_DIM 64
#define D_DIM 128
#define SLOPE 0.2f

__device__ __forceinline__ float leaky(float v) { return v >= 0.f ? v : SLOPE * v; }
__device__ __forceinline__ unsigned ord_enc(float f) {
    unsigned u = __float_as_uint(f);
    return (u >> 31) ? ~u : (u | 0x80000000u);
}
__device__ __forceinline__ float ord_dec(unsigned u) {
    return (u >> 31) ? __uint_as_float(u & 0x7FFFFFFFu) : __uint_as_float(~u);
}

// ---------------------------------------------------------------------------
// K1: per-gene linear + ReLU -> f16, stored PRE-SWIZZLED for the WMMA A operand.
// Within each 32-half K-group the order is [0..7, 16..23, 8..15, 24..31], so a
// lane's 16-half A fragment (K = kb+8*half16 .. +7 and kb+16+8*half16 .. +7) is
// one contiguous 32-byte v16h load at row*128 + ks*32 + 16*half16.
__global__ void k_gene_linear(const float* __restrict__ x,
                              const float* __restrict__ W_in,
                              const float* __restrict__ b_in,
                              _Float16* __restrict__ h_half) {
    int g = blockIdx.x;
    int d = threadIdx.x;               // 0..127
    __shared__ float xs[IN_DIM];
    if (d < IN_DIM) xs[d] = x[g * IN_DIM + d];
    __syncthreads();
    const float* W = W_in + (size_t)g * IN_DIM * D_DIM + d;
    float acc = b_in[g * D_DIM + d];
#pragma unroll 8
    for (int i = 0; i < IN_DIM; ++i) acc = fmaf(xs[i], W[(size_t)i * D_DIM], acc);
    acc = fmaxf(acc, 0.f);
    int j   = d & 31;
    int grp = j >> 3;                                       // 0..3
    int nj  = j + (grp == 1 ? 8 : (grp == 2 ? -8 : 0));     // fragment order
    h_half[(size_t)g * D_DIM + (d & ~31) + nj] = (_Float16)acc;
}

// ---------------------------------------------------------------------------
// K2: build pre-swizzled f16 B fragments from fp32 W (fuses conversion).
// Layout: [ntile(8)][kstep(4)][lane(32)][16 halfs] -> GEMM loads one v16h/lane.
// Lane l of ntile n holds column n*16+(l&15); half16=l>>4 selects K sub-ranges.
__global__ void k_bswizzle(const float* __restrict__ W, _Float16* __restrict__ Bswz) {
    int t = blockIdx.x * blockDim.x + threadIdx.x;   // 0..1023
    if (t >= 8 * 4 * 32) return;
    int lane  = t & 31;
    int kstep = (t >> 5) & 3;
    int ntile = t >> 7;
    int col = ntile * 16 + (lane & 15);
    int kb  = kstep * 32 + 8 * (lane >> 4);
    v16h frag;
#pragma unroll
    for (int e = 0; e < 8; ++e) frag[e]     = (_Float16)W[(size_t)(kb + e) * D_DIM + col];
#pragma unroll
    for (int e = 0; e < 8; ++e) frag[e + 8] = (_Float16)W[(size_t)(kb + 16 + e) * D_DIM + col];
    *(v16h*)(Bswz + (size_t)t * 16) = frag;
}

// ---------------------------------------------------------------------------
// K3: C[M,128] = A[M,128](f16,swizzled) @ B(f16,swizzled) + bias via
// v_wmma_f32_16x16x32_f16. blockDim=128 (4 waves); wave w -> N tile blockIdx.y*4+w.
// Inner loop: one v16h load for A, one v16h load for B, one WMMA per k-step.
// Store epilogue: branchless for full tiles; only the last M-tile is guarded.
__global__ void k_gemm_wmma(const _Float16* __restrict__ A,
                            const _Float16* __restrict__ Bswz,
                            const float* __restrict__ bias,
                            float* __restrict__ C, int M) {
    const int lane   = threadIdx.x & 31;
    const int wave   = threadIdx.x >> 5;
    const int m0     = blockIdx.x * 16;
    const int ntile  = blockIdx.y * 4 + wave;
    const int half16 = lane >> 4;
    const int l16    = lane & 15;

    const int row  = m0 + l16;
    const int arow = (row < M) ? row : 0;     // clamp; tail guarded on store
    const int col  = ntile * 16 + l16;

    const _Float16* Ar = A + (size_t)arow * D_DIM + 16 * half16;
    const _Float16* Br = Bswz + ((size_t)ntile * 4 * 32 + lane) * 16;

    v8f c = {};
#pragma unroll
    for (int ks = 0; ks < 4; ++ks) {
        v16h a = *(const v16h*)(Ar + ks * 32);
        v16h b = *(const v16h*)(Br + (size_t)ks * 32 * 16);
        c = __builtin_amdgcn_wmma_f32_16x16x32_f16(false, a, false, b, (short)0, c,
                                                   false, false);
    }

    const float bb = bias[col];
    float* Cp = C + (size_t)(m0 + 8 * half16) * D_DIM + col;   // row m0+8*half16, col
    if (m0 + 16 <= M) {
        // full tile: straight-line stores (312 of 313 tiles)
#pragma unroll
        for (int r = 0; r < 8; ++r) Cp[(size_t)r * D_DIM] = c[r] + bb;
    } else {
        // tail tile only
#pragma unroll
        for (int r = 0; r < 8; ++r) {
            int rr = m0 + r + 8 * half16;      // C/D: VGPR r -> row r (+8 upper lanes)
            if (rr < M) C[(size_t)rr * D_DIM + col] = c[r] + bb;
        }
    }
}

// ---------------------------------------------------------------------------
// K4: init out=bias broadcast, z=0, m=ordered(-inf), h1acc=0.
__global__ void k_init(float* __restrict__ out, const float* __restrict__ bias,
                       unsigned* __restrict__ m_ord, float* __restrict__ z,
                       float* __restrict__ h1acc, int GD, int G) {
    int i = blockIdx.x * blockDim.x + threadIdx.x;
    if (i < GD) out[i] = bias[i & (D_DIM - 1)];
    if (i < G) { m_ord[i] = 0u; z[i] = 0.f; }
    if (i < D_DIM) h1acc[i] = 0.f;
}

// ---------------------------------------------------------------------------
// K5: edge logits e_j = att . leaky(xl[src]+xr[dst]); atomic segment-max.
// One wave per edge: lane reads float4 -> full 512B row coalesced.
__global__ void k_edge_logits(const long long* __restrict__ ei,
                              const float* __restrict__ xl, const float* __restrict__ xr,
                              const float* __restrict__ att,
                              float* __restrict__ e, unsigned* __restrict__ m_ord,
                              int E, int NE) {
    int lane = threadIdx.x & 31;
    int j = blockIdx.x * (blockDim.x >> 5) + (threadIdx.x >> 5);
    if (j >= NE) return;
    int src, dst;
    if (j < E) { src = (int)ei[j]; dst = (int)ei[E + j]; }
    else       { src = dst = j - E; }

    float4 a = ((const float4*)(xl + (size_t)src * D_DIM))[lane];
    float4 b = ((const float4*)(xr + (size_t)dst * D_DIM))[lane];
    float4 t = ((const float4*)att)[lane];
    float p = t.x * leaky(a.x + b.x) + t.y * leaky(a.y + b.y) +
              t.z * leaky(a.z + b.z) + t.w * leaky(a.w + b.w);
#pragma unroll
    for (int off = 16; off > 0; off >>= 1) p += __shfl_xor(p, off, 32);
    if (lane == 0) {
        e[j] = p;
        atomicMax(&m_ord[dst], ord_enc(p));
    }
}

// ---------------------------------------------------------------------------
// K6: w_j = exp(e_j - m[dst]); atomic segment-sum into z.
__global__ void k_edge_exp(const long long* __restrict__ ei,
                           const float* __restrict__ e, const unsigned* __restrict__ m_ord,
                           float* __restrict__ w, float* __restrict__ z, int E, int NE) {
    int j = blockIdx.x * blockDim.x + threadIdx.x;
    if (j >= NE) return;
    int dst = (j < E) ? (int)ei[E + j] : (j - E);
    float wj = __expf(e[j] - ord_dec(m_ord[dst]));
    w[j] = wj;
    atomicAdd(&z[dst], wj);
}

// ---------------------------------------------------------------------------
// K7: scatter alpha * xl[src] into out[dst]. One wave per edge, float4 per lane.
__global__ void k_edge_scatter(const long long* __restrict__ ei,
                               const float* __restrict__ xl, const float* __restrict__ w,
                               const float* __restrict__ z, float* __restrict__ out,
                               int E, int NE) {
    int lane = threadIdx.x & 31;
    int j = blockIdx.x * (blockDim.x >> 5) + (threadIdx.x >> 5);
    if (j >= NE) return;
    int src, dst;
    if (j < E) { src = (int)ei[j]; dst = (int)ei[E + j]; }
    else       { src = dst = j - E; }
    float alpha = w[j] / z[dst];
    float4 v = ((const float4*)(xl + (size_t)src * D_DIM))[lane];
    float* o = out + (size_t)dst * D_DIM + lane * 4;
    atomicAdd(o + 0, alpha * v.x);
    atomicAdd(o + 1, alpha * v.y);
    atomicAdd(o + 2, alpha * v.z);
    atomicAdd(o + 3, alpha * v.w);
}

// ---------------------------------------------------------------------------
// K8: GEMV h1 += leaky(flat) @ W1, streaming 328MB of W1. Each block handles
// 512 rows; thread (col, rg) accumulates every 2nd row; atomic merge per column.
__global__ void k_mlp1(const float* __restrict__ feat, const float* __restrict__ W1,
                       float* __restrict__ h1acc, int R) {
    __shared__ float red[256];
    int col = threadIdx.x & 127;
    int rg  = threadIdx.x >> 7;                 // 0 or 1
    int r0  = blockIdx.x * 512;
    int r1  = min(r0 + 512, R);
    float acc = 0.f;
    for (int r = r0 + rg; r < r1; r += 2) {
        float f = leaky(feat[r]);
        int pr = min(r + 64, R - 1);
        __builtin_prefetch(W1 + (size_t)pr * D_DIM + col, 0, 1);
        acc = fmaf(f, W1[(size_t)r * D_DIM + col], acc);
    }
    red[threadIdx.x] = acc;
    __syncthreads();
    if (rg == 0) atomicAdd(&h1acc[col], red[col] + red[col + 128]);
}

// K9: scalar head: relu(h1+b1) @ W2 + b2 -> d_out[0].
__global__ void k_mlp2(const float* __restrict__ h1acc, const float* __restrict__ b1,
                       const float* __restrict__ W2, const float* __restrict__ b2,
                       float* __restrict__ out) {
    __shared__ float red[128];
    int t = threadIdx.x;
    red[t] = fmaxf(h1acc[t] + b1[t], 0.f) * W2[t];
    __syncthreads();
    for (int s = 64; s > 0; s >>= 1) {
        if (t < s) red[t] += red[t + s];
        __syncthreads();
    }
    if (t == 0) out[0] = red[0] + b2[0];
}

// ---------------------------------------------------------------------------
extern "C" void kernel_launch(void* const* d_in, const int* in_sizes, int n_in,
                              void* d_out, int out_size, void* d_ws, size_t ws_size,
                              hipStream_t stream) {
    const float*     x    = (const float*)d_in[0];
    const long long* ei   = (const long long*)d_in[1];   // int64 [2,E]
    const float*     W_in = (const float*)d_in[2];
    const float*     b_in = (const float*)d_in[3];
    const float*     W_l  = (const float*)d_in[4];
    const float*     b_l  = (const float*)d_in[5];
    const float*     W_r  = (const float*)d_in[6];
    const float*     b_r  = (const float*)d_in[7];
    const float*     att  = (const float*)d_in[8];
    const float*     bias = (const float*)d_in[9];
    const float*     W1   = (const float*)d_in[10];
    const float*     b1   = (const float*)d_in[11];
    const float*     W2   = (const float*)d_in[12];
    const float*     b2   = (const float*)d_in[13];
    float* out = (float*)d_out;

    const int G  = in_sizes[0] / IN_DIM;   // 5000
    const int E  = in_sizes[1] / 2;        // 240000
    const int NE = E + G;                  // + self loops
    const int GD = G * D_DIM;

    // workspace carve-out (256B aligned)
    char* ws = (char*)d_ws;
    size_t off = 0;
    auto carve = [&](size_t bytes) -> void* {
        void* p = ws + off;
        off = (off + bytes + 255) & ~(size_t)255;
        return p;
    };
    _Float16* h_half = (_Float16*)carve((size_t)GD * sizeof(_Float16));
    _Float16* Bl_swz = (_Float16*)carve((size_t)D_DIM * D_DIM * sizeof(_Float16));
    _Float16* Br_swz = (_Float16*)carve((size_t)D_DIM * D_DIM * sizeof(_Float16));
    float*    xl     = (float*)carve((size_t)GD * sizeof(float));
    float*    xr     = (float*)carve((size_t)GD * sizeof(float));
    float*    e_buf  = (float*)carve((size_t)NE * sizeof(float));
    float*    w_buf  = (float*)carve((size_t)NE * sizeof(float));
    unsigned* m_ord  = (unsigned*)carve((size_t)G * sizeof(unsigned));
    float*    z      = (float*)carve((size_t)G * sizeof(float));
    float*    outf   = (float*)carve((size_t)GD * sizeof(float));
    float*    h1acc  = (float*)carve(D_DIM * sizeof(float));

    // 1) per-gene linear + relu -> f16 (A-swizzled)
    k_gene_linear<<<G, D_DIM, 0, stream>>>(x, W_in, b_in, h_half);

    // 2) weights -> f16 fragment-swizzled B
    k_bswizzle<<<4, 256, 0, stream>>>(W_l, Bl_swz);
    k_bswizzle<<<4, 256, 0, stream>>>(W_r, Br_swz);

    // 3) WMMA GEMMs: xl = h@W_l + b_l ; xr = h@W_r + b_r
    dim3 gg((G + 15) / 16, 2);
    k_gemm_wmma<<<gg, 128, 0, stream>>>(h_half, Bl_swz, b_l, xl, G);
    k_gemm_wmma<<<gg, 128, 0, stream>>>(h_half, Br_swz, b_r, xr, G);

    // 4) init accumulators
    k_init<<<(GD + 255) / 256, 256, 0, stream>>>(outf, bias, m_ord, z, h1acc, GD, G);

    // 5-7) edge phase
    int wavesPerBlk = 256 / 32;
    k_edge_logits<<<(NE + wavesPerBlk - 1) / wavesPerBlk, 256, 0, stream>>>(
        ei, xl, xr, att, e_buf, m_ord, E, NE);
    k_edge_exp<<<(NE + 255) / 256, 256, 0, stream>>>(ei, e_buf, m_ord, w_buf, z, E, NE);
    k_edge_scatter<<<(NE + wavesPerBlk - 1) / wavesPerBlk, 256, 0, stream>>>(
        ei, xl, w_buf, z, outf, E, NE);

    // 8-9) output MLP
    k_mlp1<<<(GD + 511) / 512, 256, 0, stream>>>(outf, W1, h1acc, GD);
    k_mlp2<<<1, 128, 0, stream>>>(h1acc, b1, W2, b2, out);
}
